// AnomalyDetector_43877385896141
// MI455X (gfx1250) — matrix operand order repaired and checked by reading, not compiled, not run on hardware
//
#include <hip/hip_runtime.h>
#include <math.h>

#define B_  4
#define L_  2048
#define IN_ 64
#define D_  256
#define N_  16
#define K_  32
#define NL_ 2
#define NC_ 2
#define F_  (L_/2 + 1)   // 1025 rfft bins

typedef __attribute__((ext_vector_type(16))) _Float16 v16h;
typedef __attribute__((ext_vector_type(8)))  _Float16 v8h;
typedef __attribute__((ext_vector_type(4)))  _Float16 v4h;
typedef __attribute__((ext_vector_type(8)))  float    v8f;

enum { ACT_NONE = 0, ACT_SIGMOID = 1, ACT_SOFTPLUS = 2 };

// ---------------------------------------------------------------------------
// WMMA GEMM: C[M,N] = act(A[M,K] @ B[K,N] + bias) + resid
// f32 in memory, f16 staged in LDS in FRAGMENT-READY layouts so each lane's
// WMMA operand is a single contiguous 32B LDS read (ds_load_b128 pairs).
// Block = 128 threads (4 waves); block tile 64x16; K-step 64 (2 WMMA/step).
// Software-pipelined: global loads for tile k+1 are issued into registers
// before the WMMAs of tile k; convert+store lands in the other LDS buffer.
// K must be a multiple of 64, N of 16 (true for all uses); M may be ragged.
//
// A LDS layout: per 32-wide K subtile, columns permuted to
//   [K0..7 | K16..23 | K8..15 | K24..31]   (p = ((k&8)<<1)|((k&16)>>1)|(k&7))
// so lane (lg,lm) of wave w reads v16h at As[buf][w*16+lm][s*32 + lg*16].
// B LDS layout: transposed BsT[buf][n][k].
// ---------------------------------------------------------------------------
#define APAD 80   // row stride in halfs: 160B = mult of 32, not of 256 (bank skew)

__global__ __launch_bounds__(128)
void gemm_wmma_f16(const float* __restrict__ A, const float* __restrict__ Bm,
                   const float* __restrict__ bias, const float* __restrict__ resid,
                   float* __restrict__ C, int M, int K, int N, int act)
{
    __shared__ alignas(32) _Float16 As[2][64][APAD];
    __shared__ alignas(32) _Float16 BsT[2][16][APAD];

    const int n0   = blockIdx.x * 16;
    const int m0   = blockIdx.y * 64;
    const int t    = threadIdx.x;
    const int wave = t >> 5;
    const int lane = t & 31;
    const int lg   = lane >> 4;   // half-wave group
    const int lm   = lane & 15;

    const bool full = (m0 + 64 <= M);
    const int  cc   = t & 15;           // B-tile column handled by this thread
    const int  r8   = (t >> 4) << 3;    // B-tile k-row group

    float4 ar[8];
    float  br[8];

    // Phase 1: issue all global loads for one K-tile into registers.
    auto load_regs = [&](int k0) {
        if (full) {
            #pragma unroll
            for (int p = 0; p < 8; ++p) {
                int idx = p * 128 + t;              // float4 units of 64x64 tile
                int r   = idx >> 4;                 // 16 float4 per row
                int c4  = (idx & 15) << 2;
                ar[p] = *(const float4*)&A[(size_t)(m0 + r) * K + k0 + c4];
            }
        } else {
            #pragma unroll
            for (int p = 0; p < 8; ++p) {
                int idx = p * 128 + t;
                int r   = idx >> 4;
                int c4  = (idx & 15) << 2;
                ar[p] = make_float4(0.f, 0.f, 0.f, 0.f);
                if (m0 + r < M)
                    ar[p] = *(const float4*)&A[(size_t)(m0 + r) * K + k0 + c4];
            }
        }
        #pragma unroll
        for (int j = 0; j < 8; ++j)
            br[j] = Bm[(size_t)(k0 + r8 + j) * N + n0 + cc];
    };

    // Phase 2: convert f32->f16 and store into LDS buffer `buf`.
    auto store_lds = [&](int buf) {
        #pragma unroll
        for (int p = 0; p < 8; ++p) {
            int idx = p * 128 + t;
            int r   = idx >> 4;
            int c4  = (idx & 15) << 2;
            int kk  = c4 & 31;
            int pc  = (c4 & 32) + (((kk & 8) << 1) | ((kk & 16) >> 1) | (kk & 7));
            v4h h4; h4[0] = (_Float16)ar[p].x; h4[1] = (_Float16)ar[p].y;
                    h4[2] = (_Float16)ar[p].z; h4[3] = (_Float16)ar[p].w;
            *(v4h*)&As[buf][r][pc] = h4;
        }
        v8h hb;
        #pragma unroll
        for (int j = 0; j < 8; ++j) hb[j] = (_Float16)br[j];
        *(v8h*)&BsT[buf][cc][(r8 & 32) + ((((r8 & 31) & 8) << 1) | (((r8 & 31) & 16) >> 1))] = hb;
    };

    v8f c = {};
    const int nk  = K >> 6;
    const int row = (wave << 4) + lm;

    // prologue: stage tile 0
    load_regs(0);
    store_lds(0);
    __syncthreads();

    for (int ki = 0; ki < nk; ++ki) {
        const int  cur     = ki & 1;
        const bool hasNext = (ki + 1) < nk;
        if (hasNext) load_regs((ki + 1) << 6);      // loads in flight during WMMA

        #pragma unroll
        for (int s = 0; s < 2; ++s) {
            v16h a = *(const v16h*)&As[cur][row][(s << 5) + (lg << 4)];
            v16h b = *(const v16h*)&BsT[cur][lm][(s << 5) + (lg << 4)];
            c = __builtin_amdgcn_wmma_f32_16x16x32_f16(
                    /*neg_a=*/false, a, /*neg_b=*/false, b,
                    /*c_mod=*/(short)0, c, /*reuse_a=*/false, /*reuse_b=*/false);
        }
        if (hasNext) store_lds(cur ^ 1);
        __syncthreads();
    }

    // Epilogue: C/D 16x16 f32 layout (VGPR r: M=r lanes0-15 / M=8+r lanes16-31)
    const int col = n0 + lm;
    const float bv = bias ? bias[col] : 0.0f;
    #pragma unroll
    for (int r = 0; r < 8; ++r) {
        int orow = m0 + (wave << 4) + r + (lg << 3);
        if (orow < M) {
            float v = c[r] + bv;
            if (act == ACT_SIGMOID)       v = 1.0f / (1.0f + __expf(-v));
            else if (act == ACT_SOFTPLUS) v = (v > 20.0f) ? v : log1pf(__expf(v));
            if (resid) v += resid[(size_t)orow * N + col];
            C[(size_t)orow * N + col] = v;
        }
    }
}

// ---------------------------------------------------------------------------
// Row-wise LayerNorm over D=256 (one block per row)
// ---------------------------------------------------------------------------
__global__ __launch_bounds__(256)
void ln_rows(const float* __restrict__ X, const float* __restrict__ g,
             const float* __restrict__ bta, float* __restrict__ Y)
{
    __shared__ float red[256];
    const int row = blockIdx.x;
    const int d   = threadIdx.x;
    float x = X[(size_t)row * D_ + d];
    red[d] = x; __syncthreads();
    for (int s = 128; s > 0; s >>= 1) { if (d < s) red[d] += red[d + s]; __syncthreads(); }
    float m = red[0] * (1.0f / D_);
    __syncthreads();
    float xm = x - m;
    red[d] = xm * xm; __syncthreads();
    for (int s = 128; s > 0; s >>= 1) { if (d < s) red[d] += red[d + s]; __syncthreads(); }
    float var = red[0] * (1.0f / D_);
    Y[(size_t)row * D_ + d] = xm * rsqrtf(var + 1e-5f) * g[d] + bta[d];
}

// ---------------------------------------------------------------------------
// Depthwise conv (kernel 3, zero pad) along L
// ---------------------------------------------------------------------------
__global__ void dwconv(const float* __restrict__ X, const float* __restrict__ w,
                       const float* __restrict__ b, float* __restrict__ Y)
{
    int i = blockIdx.x * blockDim.x + threadIdx.x;
    if (i >= B_ * L_ * D_) return;
    int d = i % D_;
    int l = (i / D_) % L_;
    float xl = (l > 0)      ? X[i - D_] : 0.0f;
    float xm = X[i];
    float xr = (l < L_ - 1) ? X[i + D_] : 0.0f;
    Y[i] = xl * w[d * 3 + 0] + xm * w[d * 3 + 1] + xr * w[d * 3 + 2] + b[d];
}

// ---------------------------------------------------------------------------
// Sequential SSM scan for one batch b: one thread per (d,n) channel.
// ---------------------------------------------------------------------------
__global__ __launch_bounds__(256)
void ssm_scan(const float* __restrict__ delta, const float* __restrict__ Bt,
              const float* __restrict__ Ct, const float* __restrict__ loglam,
              float* __restrict__ ys, int b)
{
    int tid = blockIdx.x * blockDim.x + threadIdx.x;   // 0..4095
    int d = tid >> 4, n = tid & 15;
    float ll  = loglam[d * N_ + n];
    float lam = (ll > 20.0f) ? ll : log1pf(__expf(ll));   // softplus
    float h = 0.0f;
    const float* dp = delta + (size_t)b * L_ * D_ + d;
    const float* bp = Bt    + (size_t)b * L_ * N_ + n;
    const float* cp = Ct    + (size_t)b * L_ * N_ + n;
    float*       yp = ys + tid;
    for (int l = 0; l < L_; ++l) {
        float dl = dp[(size_t)l * D_];
        float a  = __expf(-dl * lam);
        h = a * h + dl * bp[(size_t)l * N_];
        yp[(size_t)l * (D_ * N_)] = h * cp[(size_t)l * N_];
    }
}

__global__ void ew_mul(const float* __restrict__ a, float* __restrict__ b, int n)
{
    int i = blockIdx.x * blockDim.x + threadIdx.x;
    if (i < n) b[i] = a[i] * b[i];
}

// ---------------------------------------------------------------------------
// DFT basis: Fr[f,t] = cos(2πft/L), Fi[f,t] = -sin(2πft/L)  (exact phase mod L)
// ---------------------------------------------------------------------------
__global__ void fill_basis(float* __restrict__ Fr, float* __restrict__ Fi)
{
    int i = blockIdx.x * blockDim.x + threadIdx.x;
    if (i >= F_ * L_) return;
    int f = i / L_, tt = i % L_;
    int ph = (int)(((long long)f * tt) & (L_ - 1));
    float ang = -6.283185307179586f * (float)ph / (float)L_;
    float s, c; sincosf(ang, &s, &c);
    Fr[i] = c; Fi[i] = s;
}

__global__ __launch_bounds__(256)
void spec_mag(const float* __restrict__ Xr, const float* __restrict__ Xi,
              float* __restrict__ mag)
{
    __shared__ float red[256];
    int f = blockIdx.x, t = threadIdx.x;
    float acc = 0.0f;
    for (int e = t; e < B_ * D_; e += 256) {
        int b = e >> 8, d = e & 255;
        size_t ix = ((size_t)b * F_ + f) * D_ + d;
        float xr = Xr[ix], xi = Xi[ix];
        acc += sqrtf(xr * xr + xi * xi);
    }
    red[t] = acc; __syncthreads();
    for (int s = 128; s > 0; s >>= 1) { if (t < s) red[t] += red[t + s]; __syncthreads(); }
    if (t == 0) mag[f] = red[0] / (float)(B_ * D_);
}

__global__ void topk32(const float* __restrict__ mag, float* __restrict__ magc,
                       int* __restrict__ idx)
{
    if (threadIdx.x != 0 || blockIdx.x != 0) return;
    for (int f = 0; f < F_; ++f) magc[f] = mag[f];
    for (int k = 0; k < K_; ++k) {
        int best = 0; float bv = -1e30f;
        for (int f = 0; f < F_; ++f) { float v = magc[f]; if (v > bv) { bv = v; best = f; } }
        idx[k] = best; magc[best] = -1e30f;
    }
}

// Filtered irfft evaluated ONLY at t = L-1 (the only row the head consumes)
__global__ void spec_last(const float* __restrict__ Xr, const float* __restrict__ Xi,
                          const int* __restrict__ idx, const float* __restrict__ fr,
                          const float* __restrict__ fi, float* __restrict__ xsl)
{
    int i = blockIdx.x * blockDim.x + threadIdx.x;  // b*D
    if (i >= B_ * D_) return;
    int b = i >> 8, d = i & 255;
    float acc = 0.0f;
    for (int k = 0; k < K_; ++k) {
        int f = idx[k];
        size_t ix = ((size_t)b * F_ + f) * D_ + d;
        float xr = Xr[ix], xi = Xi[ix];
        float wr = fr[d * K_ + k], wi = fi[d * K_ + k];
        float yr = xr * wr - xi * wi;
        float yi = xr * wi + xi * wr;
        int ph = (int)(((long long)f * (L_ - 1)) & (L_ - 1));
        float ang = 6.283185307179586f * (float)ph / (float)L_;
        float s, c; sincosf(ang, &s, &c);
        float w = (f == 0 || f == L_ / 2) ? 1.0f : 2.0f;
        acc += w * (yr * c - yi * s);
    }
    xsl[i] = acc / (float)L_;
}

// z = alpha*xt + beta*xs at t=L-1, LN, MLP head -> out (B x NC)
__global__ __launch_bounds__(256)
void head(const float* __restrict__ xt, const float* __restrict__ xsl,
          const float* __restrict__ alpha, const float* __restrict__ beta,
          const float* __restrict__ g, const float* __restrict__ bb,
          const float* __restrict__ hw1, const float* __restrict__ hb1,
          const float* __restrict__ hw2, const float* __restrict__ hb2,
          float* __restrict__ out)
{
    __shared__ float red[256];
    __shared__ float sZ[256];
    __shared__ float sH[128];
    int b = blockIdx.x, d = threadIdx.x;
    float z = alpha[d] * xt[((size_t)b * L_ + (L_ - 1)) * D_ + d] + beta[d] * xsl[b * D_ + d];
    red[d] = z; __syncthreads();
    for (int s = 128; s > 0; s >>= 1) { if (d < s) red[d] += red[d + s]; __syncthreads(); }
    float m = red[0] * (1.0f / D_);
    __syncthreads();
    float xm = z - m;
    red[d] = xm * xm; __syncthreads();
    for (int s = 128; s > 0; s >>= 1) { if (d < s) red[d] += red[d + s]; __syncthreads(); }
    float var = red[0] * (1.0f / D_);
    sZ[d] = xm * rsqrtf(var + 1e-5f) * g[d] + bb[d];
    __syncthreads();
    if (d < 128) {
        float acc = hb1[d];
        for (int e = 0; e < D_; ++e) acc += sZ[e] * hw1[e * 128 + d];
        sH[d] = acc / (1.0f + __expf(-acc));   // silu
    }
    __syncthreads();
    if (d < NC_) {
        float acc = hb2[d];
        for (int e = 0; e < 128; ++e) acc += sH[e] * hw2[e * NC_ + d];
        out[b * NC_ + d] = acc;
    }
}

// ---------------------------------------------------------------------------
static inline void launch_gemm(const float* A, const float* Bm, const float* bias,
                               const float* resid, float* C, int M, int K, int N,
                               int act, hipStream_t s)
{
    dim3 grid(N / 16, (M + 63) / 64);
    gemm_wmma_f16<<<grid, dim3(128), 0, s>>>(A, Bm, bias, resid, C, M, K, N, act);
}

extern "C" void kernel_launch(void* const* d_in, const int* in_sizes, int n_in,
                              void* d_out, int out_size, void* d_ws, size_t ws_size,
                              hipStream_t stream)
{
    const float* x      = (const float*)d_in[0];
    const float* w_in   = (const float*)d_in[1];
    const float* b_in   = (const float*)d_in[2];
    const float* ln_g   = (const float*)d_in[3];
    const float* ln_b   = (const float*)d_in[4];
    const float* conv_w = (const float*)d_in[5];
    const float* conv_b = (const float*)d_in[6];
    const float* loglam = (const float*)d_in[7];
    const float* wd     = (const float*)d_in[8];
    const float* bd     = (const float*)d_in[9];
    const float* wb     = (const float*)d_in[10];
    const float* wc     = (const float*)d_in[11];
    const float* wo     = (const float*)d_in[12];
    const float* bo     = (const float*)d_in[13];
    const float* gate_w = (const float*)d_in[14];
    const float* gate_b = (const float*)d_in[15];
    const float* proj_w = (const float*)d_in[16];
    const float* proj_b = (const float*)d_in[17];
    const float* fr     = (const float*)d_in[18];
    const float* fi     = (const float*)d_in[19];
    const float* alpha  = (const float*)d_in[20];
    const float* beta   = (const float*)d_in[21];
    const float* g_out  = (const float*)d_in[22];
    const float* b_out  = (const float*)d_in[23];
    const float* hw1    = (const float*)d_in[24];
    const float* hb1    = (const float*)d_in[25];
    const float* hw2    = (const float*)d_in[26];
    const float* hb2    = (const float*)d_in[27];

    float* W = (float*)d_ws;
    size_t off = 0;
    auto alloc = [&](size_t n) { float* p = W + off; off += n; return p; };

    const size_t BLD = (size_t)B_ * L_ * D_;
    float* h    = alloc(BLD);
    float* xt   = alloc(BLD);
    float* xn   = alloc(BLD);
    float* xc   = alloc(BLD);
    float* dl   = alloc(BLD);                      // delta
    float* Btb  = alloc((size_t)B_ * L_ * N_);
    float* Ctb  = alloc((size_t)B_ * L_ * N_);
    float* ys   = alloc((size_t)L_ * D_ * N_);     // per-batch ys (reused)
    float* t1   = alloc(BLD);                      // ssm output
    float* t2   = alloc(BLD);                      // gate, then gated ssm
    float* Fr   = alloc((size_t)F_ * L_);
    float* Fi   = alloc((size_t)F_ * L_);
    float* Xr   = alloc((size_t)B_ * F_ * D_);
    float* Xi   = alloc((size_t)B_ * F_ * D_);
    float* mag  = alloc(F_);
    float* magc = alloc(F_);
    int*   idx  = (int*)alloc(K_);
    float* xsl  = alloc((size_t)B_ * D_);

    const int BL = B_ * L_;

    // 1) input projection: h = x @ w_in + b_in
    launch_gemm(x, w_in, b_in, nullptr, h, BL, IN_, D_, ACT_NONE, stream);

    // 2) temporal blocks
    const float* xin = h;
    for (int i = 0; i < NL_; ++i) {
        ln_rows<<<BL, 256, 0, stream>>>(xin, ln_g + i * D_, ln_b + i * D_, xn);
        dwconv<<<(BL * D_ + 255) / 256, 256, 0, stream>>>(xn, conv_w + i * D_ * 3,
                                                          conv_b + i * D_, xc);
        launch_gemm(xc, wd + (size_t)i * D_ * D_, bd + i * D_, nullptr, dl,
                    BL, D_, D_, ACT_SOFTPLUS, stream);
        launch_gemm(xc, wb + (size_t)i * D_ * N_, nullptr, nullptr, Btb,
                    BL, D_, N_, ACT_NONE, stream);
        launch_gemm(xc, wc + (size_t)i * D_ * N_, nullptr, nullptr, Ctb,
                    BL, D_, N_, ACT_NONE, stream);
        for (int b = 0; b < B_; ++b) {
            ssm_scan<<<(D_ * N_) / 256, 256, 0, stream>>>(dl, Btb, Ctb,
                                                          loglam + (size_t)i * D_ * N_, ys, b);
            launch_gemm(ys, wo + (size_t)i * D_ * N_ * D_, bo + i * D_, nullptr,
                        t1 + (size_t)b * L_ * D_, L_, D_ * N_, D_, ACT_NONE, stream);
        }
        launch_gemm(xn, gate_w + (size_t)i * D_ * D_, gate_b + i * D_, nullptr, t2,
                    BL, D_, D_, ACT_SIGMOID, stream);
        ew_mul<<<(BL * D_ + 255) / 256, 256, 0, stream>>>(t1, t2, BL * D_);
        launch_gemm(t2, proj_w + (size_t)i * D_ * D_, proj_b + i * D_, xin, xt,
                    BL, D_, D_, ACT_NONE, stream);
        xin = xt;
    }

    // 3) spectral path: DFT as WMMA GEMM per batch
    fill_basis<<<(F_ * L_ + 255) / 256, 256, 0, stream>>>(Fr, Fi);
    for (int b = 0; b < B_; ++b) {
        launch_gemm(Fr, h + (size_t)b * L_ * D_, nullptr, nullptr,
                    Xr + (size_t)b * F_ * D_, F_, L_, D_, ACT_NONE, stream);
        launch_gemm(Fi, h + (size_t)b * L_ * D_, nullptr, nullptr,
                    Xi + (size_t)b * F_ * D_, F_, L_, D_, ACT_NONE, stream);
    }
    spec_mag<<<F_, 256, 0, stream>>>(Xr, Xi, mag);
    topk32<<<1, 1, 0, stream>>>(mag, magc, idx);
    spec_last<<<(B_ * D_ + 255) / 256, 256, 0, stream>>>(Xr, Xi, idx, fr, fi, xsl);

    // 4) combine + LN + MLP head (only last timestep matters)
    head<<<B_, 256, 0, stream>>>(xt, xsl, alpha, beta, g_out, b_out,
                                 hw1, hb1, hw2, hb2, (float*)d_out);
}
